// StyleProPagate_78812649881893
// MI455X (gfx1250) — matrix operand order repaired and checked by reading, not compiled, not run on hardware
//
#include <hip/hip_runtime.h>

// ---------------------------------------------------------------------------
// CDNA5 / gfx1250 implementation of the StylePropagate block.
// All heavy convs run as implicit-GEMM on v_wmma_f32_16x16x32_bf16.
// Intermediates are NHWC bf16 in workspace.
// Round 3: 2(M)x4(N) register blocking per wave (8 accumulators), tiles of
//          2 rows x 128 cols (256 pixels) -> 1.5 ds_load_b128 per WMMA.
// ---------------------------------------------------------------------------

typedef __bf16        v16bf __attribute__((ext_vector_type(16)));
typedef float         v8f   __attribute__((ext_vector_type(8)));
typedef unsigned int  v4u   __attribute__((ext_vector_type(4)));
typedef unsigned int  v2u   __attribute__((ext_vector_type(2)));

#define DEV __device__ __forceinline__

static constexpr int BATCH = 8, HH = 256, WW = 256;
static constexpr int PIXC  = 128;                 // tile columns
static constexpr int PROWS = 2;                   // tile output rows
static constexpr int PCOLS = PIXC + 2;            // halo columns in LDS patch
static constexpr int ACT_NONE = 0, ACT_LRELU = 1, ACT_RELU = 2;

DEV unsigned short f2bf(float f) {                 // fp32 -> bf16 (RNE)
  unsigned int u = __float_as_uint(f);
  u += 0x7fffu + ((u >> 16) & 1u);
  return (unsigned short)(u >> 16);
}
DEV float bf2f(unsigned short h) { return __uint_as_float(((unsigned int)h) << 16); }

union BFrag { v16bf v; v4u q[2]; };                // 32B WMMA operand fragment
union Pack8 { unsigned short u[8]; v4u q; };
union Pack4 { unsigned short u[4]; v2u q; };

// ---------------------------------------------------------------------------
// Weight prep: fp32 [CO][CI][3][3]  ->  bf16 [CO][tap=9][CIPAD] (zero padded)
// ---------------------------------------------------------------------------
__global__ void wprep_kernel(const float* __restrict__ w, unsigned short* __restrict__ dst,
                             int CO, int CI, int CIPAD) {
  int idx = blockIdx.x * 256 + threadIdx.x;
  int total = CO * 9 * CIPAD;
  if (idx >= total) return;
  int ci = idx % CIPAD; int rest = idx / CIPAD; int tap = rest % 9; int co = rest / 9;
  float v = (ci < CI) ? w[(co * CI + ci) * 9 + tap] : 0.f;
  dst[idx] = f2bf(v);
}

// ---------------------------------------------------------------------------
// featmap_in NCHW fp32 -> NHWC bf16 (LDS transpose, both sides coalesced)
// grid (W/64, H, B), 256 threads
// ---------------------------------------------------------------------------
__global__ __launch_bounds__(256) void nchw2nhwc_kernel(const float* __restrict__ in,
                                                        unsigned short* __restrict__ out) {
  __shared__ float tl[64 * 65];
  const int tid = threadIdx.x;
  const int x0 = blockIdx.x * 64, h = blockIdx.y, b = blockIdx.z;
  {
    int pix = tid & 63, cg = tid >> 6;
    for (int i = 0; i < 16; ++i) {
      int c = cg * 16 + i;
      tl[pix * 65 + c] = in[((size_t)(b * 64 + c) << 16) + h * 256 + x0 + pix];
    }
  }
  __syncthreads();
  {
    int ch = tid & 63, pg = tid >> 6;
    for (int i = 0; i < 16; ++i) {
      int p2 = pg * 16 + i;
      out[(((size_t)(b * HH + h) * WW) + x0 + p2) * 64 + ch] = f2bf(tl[p2 * 65 + ch]);
    }
  }
}

// ---------------------------------------------------------------------------
// Implicit-GEMM 3x3 conv via WMMA bf16. One workgroup = 64 Cout x 256 pixels
// (2 rows x 128 cols), weight-stationary in LDS, grid-stride over tiles of
// one batch image. 8 waves, each owning a 2(M) x 4(N) tile grid:
//   per (kc,tap) step: 2 A-frags + 4 B-frags -> 8 v_wmma (1.5 b128/WMMA).
// blockIdx.z selects between two weight sets (fused conv pairs).
// ---------------------------------------------------------------------------
template <int CIN>
__global__ __launch_bounds__(256)
void conv3x3_wmma_kernel(const unsigned short* __restrict__ Xg,   // NHWC bf16 [B,H,W,CIN]
                         const unsigned short* __restrict__ W0,   // bf16 [64][9][CIN]
                         const unsigned short* __restrict__ W1,
                         const float* __restrict__ bias0,
                         const float* __restrict__ bias1,
                         unsigned short* __restrict__ O0,         // NHWC bf16 out
                         unsigned short* __restrict__ O1,
                         int cs0, int cs1, int coff0, int coff1,
                         int act0, int act1,
                         float* __restrict__ stats_sum,           // [B*64] (z==1 only)
                         float* __restrict__ stats_sq) {
  extern __shared__ v4u smem4[];
  unsigned short* WS = (unsigned short*)smem4;                  // 64*9*CIN bf16
  unsigned short* PS = WS + 64 * 9 * CIN;                       // (PROWS+2)*PCOLS*(CIN+8)
  float* SS = (float*)(PS + (PROWS + 2) * PCOLS * (CIN + 8));   // 128 f32 stats

  const int z = blockIdx.z;
  const int b = blockIdx.y;
  const unsigned short* Wsrc = z ? W1 : W0;
  const float* bsrc = z ? bias1 : bias0;
  unsigned short* Out = z ? O1 : O0;
  const int cstride = z ? cs1 : cs0;
  const int coff = z ? coff1 : coff0;
  const int act = z ? act1 : act0;
  const bool do_stats = (z == 1) && (stats_sum != nullptr);

  const int tid = threadIdx.x;
  {                                                         // stage weights once
    const v4u* src = (const v4u*)Wsrc;
    v4u* dst = (v4u*)WS;
    const int n = 64 * 9 * CIN / 8;
    for (int i = tid; i < n; i += 256) dst[i] = src[i];
  }
  if (do_stats) for (int i = tid; i < 128; i += 256) SS[i] = 0.f;

  const int wave = tid >> 5;
  const int lane = tid & 31;
  const int ln = lane & 15;
  const int kh = lane >> 4;            // K-half selector per ISA A/B layout
  const int mt0 = (wave >> 2) * 2;     // this wave: M tiles mt0, mt0+1
  const int ng = wave & 3;             // pixel group
  const int prow = ng >> 1;            // output row within tile (0/1)
  const int pc0 = (ng & 1) * 64;       // first column of this wave's 4 N tiles
  const int CST = CIN + 8;             // padded per-pixel stride in LDS patch

  const int TILES = (HH / PROWS) * (WW / PIXC);   // 256
  for (int t = blockIdx.x; t < TILES; t += gridDim.x) {
    const int h = (t >> 1) * PROWS;
    const int x0 = (t & 1) * PIXC;
    __syncthreads();
    {                                                       // stage halo patch
      const int CCH = CIN / 8;
      const int n = (PROWS + 2) * PCOLS * CCH;
      for (int i = tid; i < n; i += 256) {
        int pix = i / CCH, cc = i - pix * CCH;
        int r = pix / PCOLS, cpos = pix - r * PCOLS;
        int hh = h + r - 1, wwp = x0 + cpos - 1;
        v4u val = {0u, 0u, 0u, 0u};
        if ((unsigned)hh < (unsigned)HH && (unsigned)wwp < (unsigned)WW)
          val = *(const v4u*)(Xg + (((size_t)(b * HH + hh) * WW + wwp) * CIN + cc * 8));
        *(v4u*)(PS + pix * CST + cc * 8) = val;
      }
    }
    if (t + (int)gridDim.x < TILES) {                       // prefetch next tile
      int t2 = t + gridDim.x;
      int h2 = (t2 >> 1) * PROWS, x2 = (t2 & 1) * PIXC;
      __builtin_prefetch(Xg + ((size_t)(b * HH + h2) * WW + x2) * CIN, 0, 1);
    }
    __syncthreads();

    v8f acc[2][4];
#pragma unroll
    for (int ii = 0; ii < 2; ++ii)
#pragma unroll
      for (int jj = 0; jj < 4; ++jj)
        acc[ii][jj] = (v8f){0.f,0.f,0.f,0.f,0.f,0.f,0.f,0.f};

    for (int kc = 0; kc < CIN / 32; ++kc) {
#pragma unroll
      for (int tap = 0; tap < 9; ++tap) {
        const int ky = tap / 3, kx = tap % 3;
        // A fragments: row m, K = kh*8+{0..7} and 16+kh*8+{0..7}
        BFrag a0, a1;
        const unsigned short* ap0 =
            WS + ((size_t)(mt0 * 16 + ln) * 9 + tap) * CIN + kc * 32 + kh * 8;
        a0.q[0] = *(const v4u*)ap0;
        a0.q[1] = *(const v4u*)(ap0 + 16);
        const unsigned short* ap1 = ap0 + (size_t)16 * 9 * CIN;
        a1.q[0] = *(const v4u*)ap1;
        a1.q[1] = *(const v4u*)(ap1 + 16);
        // B fragments: K = kh*8+{0..7} and 16+kh*8+{0..7}, column = pixel
        BFrag bf[4];
        const unsigned short* bp =
            PS + ((ky + prow) * PCOLS + pc0 + ln + kx) * CST + kc * 32 + kh * 8;
#pragma unroll
        for (int jj = 0; jj < 4; ++jj) {
          bf[jj].q[0] = *(const v4u*)bp;
          bf[jj].q[1] = *(const v4u*)(bp + 16);
          bp += 16 * CST;
        }
#pragma unroll
        for (int jj = 0; jj < 4; ++jj) {
          acc[0][jj] = __builtin_amdgcn_wmma_f32_16x16x32_bf16(
              false, a0.v, false, bf[jj].v, (short)0, acc[0][jj], false, false);
          acc[1][jj] = __builtin_amdgcn_wmma_f32_16x16x32_bf16(
              false, a1.v, false, bf[jj].v, (short)0, acc[1][jj], false, false);
        }
      }
    }

    // epilogue: bias + activation + bf16 NHWC store (+ optional stats)
#pragma unroll
    for (int ii = 0; ii < 2; ++ii) {
#pragma unroll
      for (int jj = 0; jj < 4; ++jj) {
        v8f a = acc[ii][jj];
        const int mbase = (mt0 + ii) * 16 + kh * 8;
        const int n = pc0 + jj * 16 + ln;
        const size_t pbase = (size_t)(b * HH + h + prow) * WW + x0 + n;
        float fv[8];
#pragma unroll
        for (int r = 0; r < 8; ++r) {
          float v = a[r] + bsrc[mbase + r];
          if (act == ACT_LRELU) v = (v >= 0.f) ? v : 0.2f * v;
          else if (act == ACT_RELU) v = fmaxf(v, 0.f);
          fv[r] = v;
        }
        if (do_stats) {
#pragma unroll
          for (int r = 0; r < 8; ++r) {
            atomicAdd(&SS[mbase + r], fv[r]);
            atomicAdd(&SS[64 + mbase + r], fv[r] * fv[r]);
          }
        }
        Pack8 pk;
#pragma unroll
        for (int r = 0; r < 8; ++r) pk.u[r] = f2bf(fv[r]);
        *(v4u*)(Out + pbase * cstride + coff + mbase) = pk.q;
      }
    }
  }

  if (do_stats) {
    __syncthreads();
    for (int i = tid; i < 64; i += 256) {
      atomicAdd(&stats_sum[b * 64 + i], SS[i]);
      atomicAdd(&stats_sq[b * 64 + i], SS[64 + i]);
    }
  }
}

// ---------------------------------------------------------------------------
// Masked per-class pooling: sums[b,s,f] += featmap, cnts[b,s] over pixels
// where bg>0 && mask<1 && label==s. LDS accumulate -> global atomics.
// 256 blocks * 2048 pixels.
// ---------------------------------------------------------------------------
__global__ __launch_bounds__(256)
void pool_kernel(const unsigned short* __restrict__ F, const int* __restrict__ seg,
                 const float* __restrict__ bg, const float* __restrict__ mk,
                 float* __restrict__ gsum, float* __restrict__ gcnt) {
  __shared__ float ss[35 * 64];
  __shared__ float sc[35];
  const int tid = threadIdx.x, wave = tid >> 5, lane = tid & 31;
  for (int i = tid; i < 35 * 64; i += 256) ss[i] = 0.f;
  for (int i = tid; i < 35; i += 256) sc[i] = 0.f;
  __syncthreads();
  const int base = blockIdx.x * 2048;
  const int b = base >> 16;                 // 2048 divides 65536 -> single batch
  for (int i = 0; i < 2048 / 8; ++i) {
    int p = base + i * 8 + wave;
    int lab = seg[p];
    float bgv = bg[p], mkv = mk[p];
    if (bgv > 0.f && mkv < 1.f) {
      float v0 = bf2f(F[(size_t)p * 64 + lane]);
      float v1 = bf2f(F[(size_t)p * 64 + lane + 32]);
      atomicAdd(&ss[lab * 64 + lane], v0);
      atomicAdd(&ss[lab * 64 + lane + 32], v1);
      if (lane == 0) atomicAdd(&sc[lab], 1.f);
    }
  }
  __syncthreads();
  for (int i = tid; i < 35 * 64; i += 256) atomicAdd(&gsum[b * 35 * 64 + i], ss[i]);
  for (int i = tid; i < 35; i += 256) atomicAdd(&gcnt[b * 35 + i], sc[i]);
}

// codes[b,s,f] = (cnt>0 ? sum/max(cnt,1) : 0), zero classes 24..33 -> bf16
__global__ void codes_kernel(const float* __restrict__ gsum, const float* __restrict__ gcnt,
                             unsigned short* __restrict__ codes) {
  int idx = blockIdx.x * 256 + threadIdx.x;
  if (idx >= 8 * 35 * 64) return;
  int bs = idx >> 6, s = bs % 35;
  float cnt = gcnt[bs];
  float v = (cnt > 0.f) ? gsum[idx] / fmaxf(cnt, 1.f) : 0.f;
  if (s >= 24 && s <= 33) v = 0.f;
  codes[idx] = f2bf(v);
}

// instance-norm: mu, rsigma per (b,c)
__global__ void stats_kernel(const float* __restrict__ ssum, const float* __restrict__ ssq,
                             float* __restrict__ mus, float* __restrict__ rsigs) {
  int idx = blockIdx.x * 256 + threadIdx.x;
  if (idx >= BATCH * 64) return;
  const float n = (float)(HH * WW);
  float mu = ssum[idx] / n;
  float var = ssq[idx] / n - mu * mu;
  mus[idx] = mu;
  rsigs[idx] = rsqrtf(var + 1e-5f);
}

// out_map NHWC bf16 [B,H,W,128]: ch<64 = bg*codes[label][ch];
// 64..98 = bg*(label==ch-64); 99..127 = 0.  One wave per pixel, 4 ch/lane.
__global__ __launch_bounds__(256)
void outmap_kernel(const int* __restrict__ seg, const float* __restrict__ bg,
                   const unsigned short* __restrict__ codes, unsigned short* __restrict__ om) {
  const int tid = threadIdx.x, lane = tid & 31, wv = tid >> 5;
  const size_t p = (size_t)blockIdx.x * 8 + wv;
  const int b = (int)(p >> 16);
  const int lab = seg[p];
  const float bgv = bg[p];
  const int c0 = lane * 4;
  Pack4 pk;
#pragma unroll
  for (int j = 0; j < 4; ++j) {
    int ch = c0 + j;
    float v;
    if (ch < 64) v = bgv * bf2f(codes[(b * 35 + lab) * 64 + ch]);
    else if (ch < 99) v = (ch - 64 == lab) ? bgv : 0.f;
    else v = 0.f;
    pk.u[j] = f2bf(v);
  }
  *(v2u*)(om + p * 128 + c0) = pk.q;
}

// final: out = lrelu( (dx-mu)*rsig*(1+gamma) + beta ), NHWC reads -> NCHW write
// grid (W/64, H, B)
__global__ __launch_bounds__(256)
void final_kernel(const unsigned short* __restrict__ GB, const unsigned short* __restrict__ DX,
                  const float* __restrict__ mus, const float* __restrict__ rsigs,
                  float* __restrict__ out) {
  __shared__ float tl[64 * 65];
  const int tid = threadIdx.x;
  const int x0 = blockIdx.x * 64, h = blockIdx.y, b = blockIdx.z;
  {
    int ch = tid & 63, pg = tid >> 6;
    float mu = mus[b * 64 + ch], rs = rsigs[b * 64 + ch];
    for (int i = 0; i < 16; ++i) {
      int pix = pg * 16 + i;
      size_t p = (size_t)(b * HH + h) * WW + x0 + pix;
      float dx = bf2f(DX[p * 64 + ch]);
      float nm = (dx - mu) * rs;
      float g = bf2f(GB[p * 128 + ch]);
      float be = bf2f(GB[p * 128 + 64 + ch]);
      float v = nm * (1.f + g) + be;
      tl[pix * 65 + ch] = (v >= 0.f) ? v : 0.2f * v;
    }
  }
  __syncthreads();
  {
    int pix2 = tid & 63, cg = tid >> 6;
    for (int i = 0; i < 16; ++i) {
      int c = cg * 16 + i;
      out[((size_t)(b * 64 + c) << 16) + h * 256 + x0 + pix2] = tl[pix2 * 65 + c];
    }
  }
}

// ---------------------------------------------------------------------------
extern "C" void kernel_launch(void* const* d_in, const int* in_sizes, int n_in,
                              void* d_out, int out_size, void* d_ws, size_t ws_size,
                              hipStream_t stream) {
  const float* featmap_in = (const float*)d_in[0];
  const int*   seg        = (const int*)d_in[1];
  const float* bg         = (const float*)d_in[2];
  const float* mk         = (const float*)d_in[3];
  const float* w_conv     = (const float*)d_in[4];
  const float* b_conv     = (const float*)d_in[5];
  const float* w_c0       = (const float*)d_in[6];
  const float* b_c0       = (const float*)d_in[7];
  const float* w_sh       = (const float*)d_in[8];
  const float* b_sh       = (const float*)d_in[9];
  const float* w_g        = (const float*)d_in[10];
  const float* b_g        = (const float*)d_in[11];
  const float* w_b        = (const float*)d_in[12];
  const float* b_b        = (const float*)d_in[13];
  float* out = (float*)d_out;

  const size_t NPIX = (size_t)BATCH * HH * WW;          // 524288
  char* ws = (char*)d_ws;
  size_t off = 0;
  auto alloc = [&](size_t bytes) -> void* {
    off = (off + 255) & ~(size_t)255;
    void* p = ws + off;
    off += bytes;
    return p;
  };

  // region0: X0 (NHWC bf16 64ch) | F (NHWC bf16 64ch); later aliased by ACTV(128ch)
  unsigned short* X0   = (unsigned short*)alloc(NPIX * 128 * 2);
  unsigned short* F    = X0 + NPIX * 64;
  unsigned short* ACTV = X0;                            // alias (X0,F dead by then)
  // region1: OUTMAP (128ch); later aliased by GB (gamma|beta)
  unsigned short* OM   = (unsigned short*)alloc(NPIX * 128 * 2);
  unsigned short* GB   = OM;                            // alias (OUTMAP dead by then)
  unsigned short* DX   = (unsigned short*)alloc(NPIX * 64 * 2);
  unsigned short* WC1b = (unsigned short*)alloc((size_t)64 * 9 * 64 * 2);
  unsigned short* WC0b = (unsigned short*)alloc((size_t)64 * 9 * 64 * 2);
  unsigned short* WSHb = (unsigned short*)alloc((size_t)128 * 9 * 128 * 2);
  unsigned short* WGb  = (unsigned short*)alloc((size_t)64 * 9 * 128 * 2);
  unsigned short* WBb  = (unsigned short*)alloc((size_t)64 * 9 * 128 * 2);
  float* gsum  = (float*)alloc((size_t)8 * 35 * 64 * 4);
  float* gcnt  = (float*)alloc((size_t)8 * 35 * 4);
  unsigned short* codes = (unsigned short*)alloc((size_t)8 * 35 * 64 * 2);
  float* ssum  = (float*)alloc(512 * 4);
  float* ssq   = (float*)alloc(512 * 4);
  float* mus   = (float*)alloc(512 * 4);
  float* rsigs = (float*)alloc(512 * 4);

  const dim3 blk(256);
  auto convshared = [](int cin) -> size_t {
    return (size_t)(64 * 9 * cin + (PROWS + 2) * PCOLS * (cin + 8)) * 2 + 512;
  };
  const size_t sh64 = convshared(64), sh128 = convshared(128);
  hipFuncSetAttribute((const void*)conv3x3_wmma_kernel<64>,
                      hipFuncAttributeMaxDynamicSharedMemorySize, (int)sh64);
  hipFuncSetAttribute((const void*)conv3x3_wmma_kernel<128>,
                      hipFuncAttributeMaxDynamicSharedMemorySize, (int)sh128);

  // zero accumulators
  hipMemsetAsync(gsum, 0, (size_t)8 * 35 * 64 * 4, stream);
  hipMemsetAsync(gcnt, 0, (size_t)8 * 35 * 4, stream);
  hipMemsetAsync(ssum, 0, 512 * 4, stream);
  hipMemsetAsync(ssq,  0, 512 * 4, stream);

  // weight prep (bf16, [cout][tap][cin_pad])
  wprep_kernel<<<(64 * 9 * 64 + 255) / 256, blk, 0, stream>>>(w_conv, WC1b, 64, 64, 64);
  wprep_kernel<<<(64 * 9 * 64 + 255) / 256, blk, 0, stream>>>(w_c0, WC0b, 64, 64, 64);
  wprep_kernel<<<(128 * 9 * 128 + 255) / 256, blk, 0, stream>>>(w_sh, WSHb, 128, 99, 128);
  wprep_kernel<<<(64 * 9 * 128 + 255) / 256, blk, 0, stream>>>(w_g, WGb, 64, 128, 128);
  wprep_kernel<<<(64 * 9 * 128 + 255) / 256, blk, 0, stream>>>(w_b, WBb, 64, 128, 128);

  // featmap_in -> NHWC bf16
  nchw2nhwc_kernel<<<dim3(WW / 64, HH, BATCH), blk, 0, stream>>>(featmap_in, X0);

  // fused conv1 (lrelu -> F) + conv_c0 (-> DX, instance-norm stats)
  conv3x3_wmma_kernel<64><<<dim3(128, BATCH, 2), blk, sh64, stream>>>(
      X0, WC1b, WC0b, b_conv, b_c0, F, DX, 64, 64, 0, 0,
      ACT_LRELU, ACT_NONE, ssum, ssq);

  // masked per-class pooling -> codes
  pool_kernel<<<(int)(NPIX / 2048), blk, 0, stream>>>(F, seg, bg, mk, gsum, gcnt);
  codes_kernel<<<(8 * 35 * 64 + 255) / 256, blk, 0, stream>>>(gsum, gcnt, codes);
  stats_kernel<<<2, blk, 0, stream>>>(ssum, ssq, mus, rsigs);

  // out_map = [style_map | bg_segmap | zero-pad]
  outmap_kernel<<<(int)(NPIX / 8), blk, 0, stream>>>(seg, bg, codes, OM);

  // SPADE shared conv (99->128 padded, relu), split Cout across z
  conv3x3_wmma_kernel<128><<<dim3(128, BATCH, 2), blk, sh128, stream>>>(
      OM, WSHb, WSHb + (size_t)64 * 9 * 128, b_sh, b_sh + 64, ACTV, ACTV,
      128, 128, 0, 64, ACT_RELU, ACT_RELU, nullptr, nullptr);

  // gamma (z=0) and beta (z=1) convs fused into GB [B,H,W,128]
  conv3x3_wmma_kernel<128><<<dim3(128, BATCH, 2), blk, sh128, stream>>>(
      ACTV, WGb, WBb, b_g, b_b, GB, GB, 128, 128, 0, 64,
      ACT_NONE, ACT_NONE, nullptr, nullptr);

  // final: instance-norm affine + lrelu -> NCHW fp32 output
  final_kernel<<<dim3(WW / 64, HH, BATCH), blk, 0, stream>>>(GB, DX, mus, rsigs, out);
}
